// LSTM_44083544326519
// MI455X (gfx1250) — matrix operand order repaired and checked by reading, not compiled, not run on hardware
//
#include <hip/hip_runtime.h>

typedef __attribute__((ext_vector_type(16))) __bf16 v16bf;
typedef __attribute__((ext_vector_type(8)))  float  v8f;

#define B_  256
#define T_  512
#define I_  256
#define H_  512
#define K_  768            // I + H
#define NG  2048           // 4 * H  (gate columns: f | i | g | o)
#define KT  24             // K / 32
#define AP  776            // padded A row stride in bf16 elements (768 + 8)
#define WPACK_ELEMS (KT * 128 * 32 * 16)   // 1,572,864 bf16 = 3 MB

__device__ __forceinline__ unsigned short f2bf(float x) {
    union { float f; unsigned u; } v; v.f = x;
    unsigned u = v.u;
    return (unsigned short)((u + 0x7FFFu + ((u >> 16) & 1u)) >> 16); // RNE
}

__device__ __forceinline__ float sigm(float x) {
    return 1.0f / (1.0f + __expf(-x));
}
__device__ __forceinline__ float tanh_fast(float x) {
    return 2.0f / (1.0f + __expf(-2.0f * x)) - 1.0f;
}

// ---------------------------------------------------------------------------
// Pack kernel: W_all (768 x 2048, fp32, columns = [f|i|g|o]) -> bf16 in the
// per-lane B-operand layout used here (contiguous k, same order as A):
//   Wpack[((kt*128 + nt)*32 + lane)*16 + j] = W_all[kt*32 + (lane>>4)*16 + j]
//                                                  [nt*16 + (lane&15)]
// Also packs the 4 bias vectors into b_all[2048].
// ---------------------------------------------------------------------------
__global__ void lstm_pack(const float* __restrict__ Wf, const float* __restrict__ Wi,
                          const float* __restrict__ Wc, const float* __restrict__ Wo,
                          const float* __restrict__ bfv, const float* __restrict__ biv,
                          const float* __restrict__ bcv, const float* __restrict__ bov,
                          unsigned short* __restrict__ Wpack, float* __restrict__ b_all) {
    int idx = blockIdx.x * blockDim.x + threadIdx.x;
    if (idx >= WPACK_ELEMS) return;
    int j    = idx & 15;
    int lane = (idx >> 4) & 31;
    int nt   = (idx >> 9) & 127;
    int kt   = idx >> 16;
    int k = kt * 32 + ((lane >> 4) << 4) + j;
    int n = (nt << 4) + (lane & 15);
    int gate = n >> 9;
    int h    = n & (H_ - 1);
    const float* W = (gate == 0) ? Wf : (gate == 1) ? Wi : (gate == 2) ? Wc : Wo;
    Wpack[idx] = f2bf(W[k * H_ + h]);
    if (idx < NG) {
        int g2 = idx >> 9, h2 = idx & (H_ - 1);
        const float* bb = (g2 == 0) ? bfv : (g2 == 1) ? biv : (g2 == 2) ? bcv : bov;
        b_all[idx] = bb[h2];
    }
}

// ---------------------------------------------------------------------------
// Recurrent kernel: 16 workgroups (one per 16-row batch tile), 16 waves each
// (512 threads -> 4 waves per SIMD for latency hiding).
// Wave w owns gate columns {g*512 + [w*32, (w+1)*32)} for all 4 gates, so the
// gate nonlinearity + c/h update stays entirely in accumulator registers.
// B operands are double-buffered in registers (prefetch distance = 1 K-tile).
// A operand (x_t | h_{t-1}) is staged in LDS as bf16, row stride AP=776.
// ---------------------------------------------------------------------------
__global__ void __launch_bounds__(512, 1) lstm_recurrent(
        const float* __restrict__ inputs,            // (B, T, I) fp32
        const unsigned short* __restrict__ Wpack,    // packed bf16 weights
        const float* __restrict__ b_all,             // 2048 biases
        float* __restrict__ h_last) {                // (B, H) fp32 out
    __shared__ __align__(16) unsigned short Abuf[16 * AP];

    const int tid   = threadIdx.x;
    const int wave  = tid >> 5;          // 0..15
    const int lane  = tid & 31;
    const int b0    = blockIdx.x << 4;
    const int lhalf = lane >> 4;         // 0 / 1
    const int l15   = lane & 15;

    // zero A buffer (h part must be zero for t = 0)
    for (int i = tid; i < 16 * AP; i += 512) Abuf[i] = 0;

    // per-lane biases: index gs = gate*2 + slot
    float bias[8];
    #pragma unroll
    for (int gs = 0; gs < 8; ++gs) {
        int g = gs >> 1, s = gs & 1;
        bias[gs] = b_all[g * H_ + ((wave * 2 + s) << 4) + l15];
    }

    v8f cst[2] = {};                     // cell state, WMMA C/D layout

    const unsigned short* wp = Wpack + (lane << 4);
    const int MA = l15;                  // A-operand row for this lane
    const int c0 = lhalf << 4;           // A-operand K sub-offset (0 or 16)
    const int xr = tid >> 5;             // x staging: row (0..15)
    const int xc = (tid & 31) << 3;      // x staging: col base (8 floats each)

    union BT { v16bf v; uint4 q[2]; };

    // load 8 B tiles (gs = gate*2 + slot) for K-tile ktv into buf
    auto loadB = [&](BT* buf, int ktv) {
        #pragma unroll
        for (int gs = 0; gs < 8; ++gs) {
            int nt = ((gs >> 1) << 5) + (wave << 1) + (gs & 1);
            const unsigned short* bp = wp + (((ktv << 7) + nt) << 9);
            buf[gs].q[0] = *(const uint4*)(bp);
            buf[gs].q[1] = *(const uint4*)(bp + 8);
        }
    };
    // load this lane's A-operand slice for K-tile ktv from LDS
    auto loadA = [&](int ktv) {
        BT a;
        const unsigned short* ap = &Abuf[MA * AP + ktv * 32 + c0];
        a.q[0] = *(const uint4*)(ap);
        a.q[1] = *(const uint4*)(ap + 8);
        return a;
    };

    __syncthreads();

    for (int t = 0; t < T_; ++t) {
        // ---- stage x_t tile (16 x 256) into A columns [0, 256) as bf16 ----
        {
            const float* src = inputs + (((size_t)(b0 + xr) * T_ + t) * I_) + xc;
            unsigned short* dst = &Abuf[xr * AP + xc];
            #pragma unroll
            for (int j2 = 0; j2 < 8; ++j2) dst[j2] = f2bf(src[j2]);
        }
        __syncthreads();

        // ---- GEMM with register double-buffered B operands ----
        v8f acc[8] = {};
        BT bb0[8], bb1[8];
        loadB(bb0, 0);
        for (int kt2 = 0; kt2 < 12; ++kt2) {
            const int kta = kt2 * 2, ktb = kta + 1;

            BT a0 = loadA(kta);
            loadB(bb1, ktb);                 // prefetch next K-tile
            #pragma unroll
            for (int gs = 0; gs < 8; ++gs)
                acc[gs] = __builtin_amdgcn_wmma_f32_16x16x32_bf16(
                    false, a0.v, false, bb0[gs].v, (short)0, acc[gs], false, false);

            BT a1 = loadA(ktb);
            loadB(bb0, ktb + 1);             // kt=24 prefetch lands in-ws, unused
            #pragma unroll
            for (int gs = 0; gs < 8; ++gs)
                acc[gs] = __builtin_amdgcn_wmma_f32_16x16x32_bf16(
                    false, a1.v, false, bb1[gs].v, (short)0, acc[gs], false, false);
        }
        __syncthreads();   // all waves done reading A for this step

        // ---- gates + state update, entirely in registers ----
        #pragma unroll
        for (int s = 0; s < 2; ++s) {
            #pragma unroll
            for (int r = 0; r < 8; ++r) {
                float fg = sigm(     acc[0 + s][r] + bias[0 + s]);
                float ig = sigm(     acc[2 + s][r] + bias[2 + s]);
                float gg = tanh_fast(acc[4 + s][r] + bias[4 + s]);
                float og = sigm(     acc[6 + s][r] + bias[6 + s]);
                float c  = fg * cst[s][r] + ig * gg;
                cst[s][r] = c;
                float h  = og * tanh_fast(c);
                int row  = r + (lhalf << 3);
                int nh   = ((wave * 2 + s) << 4) + l15;
                Abuf[row * AP + I_ + nh] = f2bf(h);
                if (t == T_ - 1)
                    h_last[(size_t)(b0 + row) * H_ + nh] = h;
            }
        }
        __syncthreads();
    }
}

// ---------------------------------------------------------------------------
// Output layer: logits = h_last @ Wout + bout, then log_softmax per row.
// One block per batch row, one thread per output column.
// ---------------------------------------------------------------------------
__global__ void lstm_out(const float* __restrict__ h_last,   // (B, H)
                         const float* __restrict__ Wout,     // (H, O)
                         const float* __restrict__ bout,     // (O)
                         float* __restrict__ out) {          // (B, O)
    __shared__ float hrow[H_];
    __shared__ float red[256];
    const int b = blockIdx.x, tid = threadIdx.x;

    hrow[tid]       = h_last[(size_t)b * H_ + tid];
    hrow[tid + 256] = h_last[(size_t)b * H_ + 256 + tid];
    __syncthreads();

    float acc = bout[tid];
    #pragma unroll 8
    for (int k = 0; k < H_; ++k)
        acc = fmaf(hrow[k], Wout[k * 256 + tid], acc);

    // max reduction
    red[tid] = acc; __syncthreads();
    for (int s = 128; s > 0; s >>= 1) {
        if (tid < s) red[tid] = fmaxf(red[tid], red[tid + s]);
        __syncthreads();
    }
    float m = red[0];
    __syncthreads();

    // sum(exp) reduction
    red[tid] = __expf(acc - m); __syncthreads();
    for (int s = 128; s > 0; s >>= 1) {
        if (tid < s) red[tid] += red[tid + s];
        __syncthreads();
    }
    float lse = m + __logf(red[0]);
    out[(size_t)b * 256 + tid] = acc - lse;
}

// ---------------------------------------------------------------------------
extern "C" void kernel_launch(void* const* d_in, const int* in_sizes, int n_in,
                              void* d_out, int out_size, void* d_ws, size_t ws_size,
                              hipStream_t stream) {
    const float* inputs = (const float*)d_in[0];
    const float* Wf   = (const float*)d_in[1];
    const float* bfv  = (const float*)d_in[2];
    const float* Wi   = (const float*)d_in[3];
    const float* biv  = (const float*)d_in[4];
    const float* Wc   = (const float*)d_in[5];
    const float* bcv  = (const float*)d_in[6];
    const float* Wo   = (const float*)d_in[7];
    const float* bov  = (const float*)d_in[8];
    const float* Wout = (const float*)d_in[9];
    const float* bout = (const float*)d_in[10];

    unsigned short* Wpack = (unsigned short*)d_ws;                     // 3 MB
    float* b_all  = (float*)((char*)d_ws + (size_t)WPACK_ELEMS * 2);   // 8 KB
    float* h_last = (float*)((char*)d_ws + (size_t)WPACK_ELEMS * 2 + NG * 4); // 512 KB

    lstm_pack<<<WPACK_ELEMS / 256, 256, 0, stream>>>(
        Wf, Wi, Wc, Wo, bfv, biv, bcv, bov, Wpack, b_all);

    lstm_recurrent<<<B_ / 16, 512, 0, stream>>>(inputs, Wpack, b_all, h_last);

    lstm_out<<<B_, 256, 0, stream>>>(h_last, Wout, bout, (float*)d_out);
}
